// MultiHeadSelfAttention_43731357008344
// MI455X (gfx1250) — matrix operand order repaired and checked by reading, not compiled, not run on hardware
//
#include <hip/hip_runtime.h>
#include <hip/hip_bf16.h>
#include <stdint.h>

// ---------------------------------------------------------------------------
// MHA for B=4, L=2048, D=1024, H=16, Dh=64 on gfx1250 (CDNA5, wave32, WMMA).
// Pipeline: f32->bf16 convert -> WMMA GEMM (Q,K,V) [async-LDS double-buffered]
//           -> flash-attn (WMMA, K/V from global, P via LDS) -> WMMA GEMM out-proj (f32).
// ---------------------------------------------------------------------------

typedef __attribute__((ext_vector_type(16))) __bf16 v16bf;
typedef __attribute__((ext_vector_type(8)))  __bf16 v8bf;
typedef __attribute__((ext_vector_type(8)))  float  v8f;

#define D_MODEL 1024
#define SEQ_L   2048
#define NBATCH  4
#define NHEADS  16
#define DHEAD   64
#define MTOT    (NBATCH * SEQ_L)   // 8192

// round-to-nearest-even f32 -> bf16 (bits)
__device__ __forceinline__ unsigned short f2bf(float f) {
  unsigned u = __float_as_uint(f);
  unsigned r = u + 0x7FFFu + ((u >> 16) & 1u);
  return (unsigned short)(r >> 16);
}

__device__ __forceinline__ v8f wmma_bf16(v16bf a, v16bf b, v8f c) {
  return __builtin_amdgcn_wmma_f32_16x16x32_bf16(
      /*neg_a=*/false, a, /*neg_b=*/false, b,
      /*c_mod=*/(short)0, c, /*reuse_a=*/false, /*reuse_b=*/false);
}

// assemble one 16x32 A / 32x16 B fragment lane payload from two 16B pieces
__device__ __forceinline__ v16bf load_frag2(const unsigned short* p0,
                                            const unsigned short* p1) {
  union { v16bf v; v8bf h[2]; } u;
  u.h[0] = *(const v8bf*)p0;
  u.h[1] = *(const v8bf*)p1;
  return u.v;
}

__device__ __forceinline__ v8f vzero8() {
  v8f z = {0.f, 0.f, 0.f, 0.f, 0.f, 0.f, 0.f, 0.f};
  return z;
}

// CDNA5 async global->LDS 16B copy (per-lane). LDS byte address in a VGPR,
// 64-bit global address in a VGPR pair. Tracked by ASYNCcnt.
__device__ __forceinline__ void async_ld16(unsigned lds_addr,
                                           const unsigned short* g) {
  asm volatile("global_load_async_to_lds_b128 %0, %1, off"
               :: "v"(lds_addr), "v"((unsigned long long)(uintptr_t)g)
               : "memory");
}

// ---------------------------------------------------------------------------
// fp32 -> bf16 convert
// ---------------------------------------------------------------------------
__global__ void f32_to_bf16_k(const float* __restrict__ in,
                              unsigned short* __restrict__ out, int n) {
  int i = blockIdx.x * blockDim.x + threadIdx.x;
  int stride = gridDim.x * blockDim.x;
  for (; i < n; i += stride) out[i] = f2bf(in[i]);
}

// ---------------------------------------------------------------------------
// bf16 GEMM:  Y[m,n] = sum_k A[m,k] * W[n,k]   (A: [M][1024], W: [1024][1024])
// Block tile 128x64, K-step 64, double-buffered LDS filled by
// GLOBAL_LOAD_ASYNC_TO_LDS_B128; 8 waves in 4(M)x2(N), each wave 32x32.
// OUT_MODE 0: bf16 out, layout [B,H,L,Dh]      (Q, K)
// OUT_MODE 1: bf16 out, layout [B,H,Dh,L]      (V transposed)
// OUT_MODE 2: f32  out, row-major [M][1024]    (final projection)
// ---------------------------------------------------------------------------
#define BM 128
#define BN 64
#define BK 64
#define LDSA 72   // 64 + 8 pad, rows stay 16B aligned (144B stride)
#define LDSB 72
#define NKSTEP (D_MODEL / BK)   // 16

template <int OUT_MODE>
__global__ __launch_bounds__(256) void gemm_bf16_k(
    const unsigned short* __restrict__ A,
    const unsigned short* __restrict__ W,
    void* __restrict__ outp) {
  __shared__ unsigned short As[2][BM * LDSA];   // 2 x 18 KB
  __shared__ unsigned short Bs[2][BN * LDSB];   // 2 x  9 KB

  const int tid  = threadIdx.x;
  const int lane = tid & 31;
  const int wave = tid >> 5;
  const int wm   = wave & 3;   // M slice (x32)
  const int wn   = wave >> 2;  // N slice (x32)
  const int hl   = lane >> 4;  // half-wave
  const int l15  = lane & 15;

  const int m0 = blockIdx.y * BM;
  const int n0 = blockIdx.x * BN;
  const int K = D_MODEL, N = D_MODEL;

  v8f acc[2][2];
  acc[0][0] = vzero8(); acc[0][1] = vzero8();
  acc[1][0] = vzero8(); acc[1][1] = vzero8();

  // issue one block tile (A: 128x64 = 1024 16B chunks; B: 64x64 = 512 chunks)
  // -> 6 async 16B copies per thread per tile
  auto issue_tile = [&](int buf, int k0) {
#pragma unroll
    for (int i = 0; i < 4; ++i) {
      int c = tid + i * 256;
      int row = c >> 3, ch = c & 7;
      async_ld16((unsigned)(uintptr_t)&As[buf][row * LDSA + ch * 8],
                 A + (size_t)(m0 + row) * K + k0 + ch * 8);
    }
#pragma unroll
    for (int i = 0; i < 2; ++i) {
      int c = tid + i * 256;
      int row = c >> 3, ch = c & 7;
      async_ld16((unsigned)(uintptr_t)&Bs[buf][row * LDSB + ch * 8],
                 W + (size_t)(n0 + row) * K + k0 + ch * 8);
    }
  };

  issue_tile(0, 0);

  for (int it = 0; it < NKSTEP; ++it) {
    const int cur = it & 1;
    if (it + 1 < NKSTEP) {
      issue_tile(cur ^ 1, (it + 1) * BK);
      // my 6 oldest (current tile) async copies done; 6 newest may be in flight
      asm volatile("s_wait_asynccnt 0x6" ::: "memory");
    } else {
      asm volatile("s_wait_asynccnt 0x0" ::: "memory");
    }
    __syncthreads();   // tile `cur` visible to all waves

#pragma unroll
    for (int ks = 0; ks < 2; ++ks) {   // two K=32 sub-steps
      v16bf af[2], bfr[2];
#pragma unroll
      for (int ms = 0; ms < 2; ++ms) {
        // A lane layout: row = lane%16; half0 -> k 0..15, half1 -> k 16..31
        const unsigned short* base =
            As[cur] + (wm * 32 + ms * 16 + l15) * LDSA + ks * 32 + hl * 16;
        af[ms] = load_frag2(base, base + 8);
      }
#pragma unroll
      for (int ns = 0; ns < 2; ++ns) {
        // B lane layout: col = lane%16; half0 -> k 0..15, half1 -> k 16..31
        const unsigned short* base =
            Bs[cur] + (wn * 32 + ns * 16 + l15) * LDSB + ks * 32 + hl * 16;
        bfr[ns] = load_frag2(base, base + 8);
      }
#pragma unroll
      for (int ms = 0; ms < 2; ++ms)
#pragma unroll
        for (int ns = 0; ns < 2; ++ns)
          acc[ms][ns] = wmma_bf16(af[ms], bfr[ns], acc[ms][ns]);
    }
    __syncthreads();   // reads done before buffer `cur` is refilled next+1
  }

  // store (C layout: VGPR v -> row v + 8*hl, col = lane%16)
#pragma unroll
  for (int ms = 0; ms < 2; ++ms)
#pragma unroll
    for (int ns = 0; ns < 2; ++ns)
#pragma unroll
      for (int v = 0; v < 8; ++v) {
        int row = m0 + wm * 32 + ms * 16 + v + hl * 8;
        int col = n0 + wn * 32 + ns * 16 + l15;
        float val = acc[ms][ns][v];
        if (OUT_MODE == 2) {
          ((float*)outp)[(size_t)row * N + col] = val;
        } else {
          int b = row >> 11, l = row & (SEQ_L - 1);
          int h = col >> 6, d = col & (DHEAD - 1);
          if (OUT_MODE == 0) {  // [B,H,L,Dh]
            ((unsigned short*)outp)[((((size_t)b * NHEADS + h) * SEQ_L + l) << 6) + d] =
                f2bf(val);
          } else {              // [B,H,Dh,L]
            ((unsigned short*)outp)[(((size_t)b * NHEADS + h) * DHEAD + d) * SEQ_L + l] =
                f2bf(val);
          }
        }
      }
}

// ---------------------------------------------------------------------------
// Flash attention: 1 wave (32 threads) per 16-query tile per (b,h).
// Online softmax over key tiles of 32. All matmuls via WMMA bf16.
// K/V fragments load straight from global (working set lives in 192MB L2).
// ---------------------------------------------------------------------------
__global__ __launch_bounds__(32) void attn_k(
    const unsigned short* __restrict__ Qb,   // [B,H,L,64] bf16
    const unsigned short* __restrict__ Kb,   // [B,H,L,64] bf16
    const unsigned short* __restrict__ Vt,   // [B,H,64,L] bf16
    unsigned short* __restrict__ ctxb) {     // [B*L, 1024] bf16
  __shared__ unsigned short Pld[16 * 40];    // P tile bounce (C->A layout)

  const int lane = threadIdx.x & 31;
  const int hl   = lane >> 4;
  const int l15  = lane & 15;

  const int bid = blockIdx.x;
  const int qt  = bid & (SEQ_L / 16 - 1);  // 0..127
  const int bh  = bid >> 7;                // 0..63
  const int b   = bh >> 4;
  const int h   = bh & 15;
  const int q0  = qt * 16;

  const unsigned short* Qbase = Qb + ((size_t)bh * SEQ_L + q0) * DHEAD;
  const unsigned short* Kbase = Kb + (size_t)bh * SEQ_L * DHEAD;
  const unsigned short* Vbase = Vt + (size_t)bh * DHEAD * SEQ_L;

  // Q A-fragments: row = lane%16, d-chunks 0..31 and 32..63
  const unsigned short* qr = Qbase + (size_t)l15 * DHEAD;
  v16bf aq0 = load_frag2(qr + hl * 16,      qr + hl * 16 + 8);
  v16bf aq1 = load_frag2(qr + 32 + hl * 16, qr + 32 + hl * 16 + 8);

  v8f ctx[4];
#pragma unroll
  for (int f = 0; f < 4; ++f) ctx[f] = vzero8();
  float mst[8], lst[8];
#pragma unroll
  for (int v = 0; v < 8; ++v) { mst[v] = -1e30f; lst[v] = 0.f; }

  for (int kt = 0; kt < SEQ_L; kt += 32) {
    // K B-fragments (keys kt..kt+15 and kt+16..kt+31), contiguous per lane
    const unsigned short* kr0 = Kbase + (size_t)(kt + l15) * DHEAD;
    const unsigned short* kr1 = kr0 + 16 * DHEAD;
    v16bf b0a = load_frag2(kr0 + hl * 16,      kr0 + hl * 16 + 8);   // d 0..31
    v16bf b0b = load_frag2(kr0 + 32 + hl * 16, kr0 + 32 + hl * 16 + 8);
    v16bf b1a = load_frag2(kr1 + hl * 16,      kr1 + hl * 16 + 8);
    v16bf b1b = load_frag2(kr1 + 32 + hl * 16, kr1 + 32 + hl * 16 + 8);

    v8f s0 = wmma_bf16(aq1, b0b, wmma_bf16(aq0, b0a, vzero8()));
    v8f s1 = wmma_bf16(aq1, b1b, wmma_bf16(aq0, b1a, vzero8()));

    float alpha[8];
#pragma unroll
    for (int v = 0; v < 8; ++v) {
      float x0 = s0[v] * 0.125f;   // 1/sqrt(64)
      float x1 = s1[v] * 0.125f;
      float mx = fmaxf(x0, x1);
      mx = fmaxf(mx, __shfl_xor(mx, 1, 32));
      mx = fmaxf(mx, __shfl_xor(mx, 2, 32));
      mx = fmaxf(mx, __shfl_xor(mx, 4, 32));
      mx = fmaxf(mx, __shfl_xor(mx, 8, 32));
      float mn = fmaxf(mst[v], mx);
      float e0 = __expf(x0 - mn);
      float e1 = __expf(x1 - mn);
      s0[v] = e0; s1[v] = e1;
      float rs = e0 + e1;
      rs += __shfl_xor(rs, 1, 32);
      rs += __shfl_xor(rs, 2, 32);
      rs += __shfl_xor(rs, 4, 32);
      rs += __shfl_xor(rs, 8, 32);
      float a = __expf(mst[v] - mn);
      mst[v] = mn;
      lst[v] = lst[v] * a + rs;
      alpha[v] = a;
    }
#pragma unroll
    for (int f = 0; f < 4; ++f)
#pragma unroll
      for (int v = 0; v < 8; ++v) ctx[f][v] *= alpha[v];

    // P (C layout) -> LDS -> A layout
#pragma unroll
    for (int v = 0; v < 8; ++v) {
      Pld[(v + hl * 8) * 40 + l15]      = f2bf(s0[v]);
      Pld[(v + hl * 8) * 40 + 16 + l15] = f2bf(s1[v]);
    }
    __syncthreads();
    const unsigned short* pr = Pld + l15 * 40 + hl * 16;
    v16bf ap = load_frag2(pr, pr + 8);

    // ctx += P (16x32) x V (32x64): V B-fragments straight from transposed V
#pragma unroll
    for (int f = 0; f < 4; ++f) {
      const unsigned short* vr =
          Vbase + (size_t)(f * 16 + l15) * SEQ_L + kt + hl * 16;
      v16bf bv = load_frag2(vr, vr + 8);
      ctx[f] = wmma_bf16(ap, bv, ctx[f]);
    }
    __syncthreads();
  }

  float inv[8];
#pragma unroll
  for (int v = 0; v < 8; ++v) inv[v] = 1.0f / lst[v];

  const size_t rowBase = (size_t)b * SEQ_L + q0;
#pragma unroll
  for (int f = 0; f < 4; ++f)
#pragma unroll
    for (int v = 0; v < 8; ++v) {
      float val = ctx[f][v] * inv[v];
      ctxb[(rowBase + v + hl * 8) * D_MODEL + h * DHEAD + f * 16 + l15] =
          f2bf(val);
    }
}

// ---------------------------------------------------------------------------
// launch
// ---------------------------------------------------------------------------
extern "C" void kernel_launch(void* const* d_in, const int* in_sizes, int n_in,
                              void* d_out, int out_size, void* d_ws, size_t ws_size,
                              hipStream_t stream) {
  const float* x  = (const float*)d_in[0];
  const float* Wq = (const float*)d_in[1];
  const float* Wk = (const float*)d_in[2];
  const float* Wv = (const float*)d_in[3];
  const float* Wo = (const float*)d_in[4];

  char* ws = (char*)d_ws;
  size_t off = 0;
  unsigned short* xb  = (unsigned short*)(ws + off); off += (size_t)MTOT * D_MODEL * 2;
  unsigned short* Wqb = (unsigned short*)(ws + off); off += (size_t)D_MODEL * D_MODEL * 2;
  unsigned short* Wkb = (unsigned short*)(ws + off); off += (size_t)D_MODEL * D_MODEL * 2;
  unsigned short* Wvb = (unsigned short*)(ws + off); off += (size_t)D_MODEL * D_MODEL * 2;
  unsigned short* Wob = (unsigned short*)(ws + off); off += (size_t)D_MODEL * D_MODEL * 2;
  unsigned short* Qb  = (unsigned short*)(ws + off); off += (size_t)MTOT * D_MODEL * 2;
  unsigned short* Kb  = (unsigned short*)(ws + off); off += (size_t)MTOT * D_MODEL * 2;
  unsigned short* Vtb = (unsigned short*)(ws + off); off += (size_t)MTOT * D_MODEL * 2;
  unsigned short* ctxb= (unsigned short*)(ws + off); off += (size_t)MTOT * D_MODEL * 2;
  (void)ws_size; // requires ~92 MB of workspace

  const int nX = MTOT * D_MODEL;
  const int nW = D_MODEL * D_MODEL;
  f32_to_bf16_k<<<1024, 256, 0, stream>>>(x,  xb,  nX);
  f32_to_bf16_k<<<512,  256, 0, stream>>>(Wq, Wqb, nW);
  f32_to_bf16_k<<<512,  256, 0, stream>>>(Wk, Wkb, nW);
  f32_to_bf16_k<<<512,  256, 0, stream>>>(Wv, Wvb, nW);
  f32_to_bf16_k<<<512,  256, 0, stream>>>(Wo, Wob, nW);

  dim3 ggrid(D_MODEL / BN, MTOT / BM);  // (16, 64)
  gemm_bf16_k<0><<<ggrid, 256, 0, stream>>>(xb, Wqb, Qb);
  gemm_bf16_k<0><<<ggrid, 256, 0, stream>>>(xb, Wkb, Kb);
  gemm_bf16_k<1><<<ggrid, 256, 0, stream>>>(xb, Wvb, Vtb);

  attn_k<<<NBATCH * NHEADS * (SEQ_L / 16), 32, 0, stream>>>(Qb, Kb, Vtb, ctxb);

  gemm_bf16_k<2><<<ggrid, 256, 0, stream>>>(ctxb, Wob, d_out);
  (void)in_sizes; (void)n_in; (void)out_size;
}
